// CausalAttention_59322088293055
// MI455X (gfx1250) — matrix operand order repaired and checked by reading, not compiled, not run on hardware
//
#include <hip/hip_runtime.h>
#include <hip/hip_bf16.h>

typedef __attribute__((ext_vector_type(16))) _Float16 v16h;
typedef __attribute__((ext_vector_type(8)))  _Float16 v8h;
typedef __attribute__((ext_vector_type(8)))  float    v8f;

#define B_  2
#define S_  2048
#define D_  1024
#define H_  16
#define DH_ 64

// ---- WMMA fragment loaders (16x16x32 f16, contiguous k order shared by A and B) ----
// A (16x32, MxK): lane m=lane&15, h=lane>>4; halves[0..15]=A[m][h*16+i]
__device__ __forceinline__ v16h frag_a(const _Float16* base, int ld, int lane) {
  const int m = lane & 15, hf = lane >> 4;
  return *(const v16h*)(base + m * ld + hf * 16);
}
// B (32x16, KxN) from row-major Bt[n][k]: lane n=lane&15 holds 16 contiguous K at (lane>>4)*16
__device__ __forceinline__ v16h frag_b(const _Float16* baseT, int ld, int lane) {
  const int n = lane & 15, hf = lane >> 4;
  return *(const v16h*)(baseT + n * ld + hf * 16);
}
__device__ __forceinline__ v8f wmma_f16(v16h a, v16h b, v8f c) {
  return __builtin_amdgcn_wmma_f32_16x16x32_f16(false, a, false, b, (short)0, c, false, false);
}

// ---- 1) x (f32) -> f16 ----
__global__ void k_cvt_x(const float* __restrict__ x, _Float16* __restrict__ xh) {
  const int i = (blockIdx.x * 256 + threadIdx.x) * 4;
  float4 v = *(const float4*)(x + i);
  xh[i + 0] = (_Float16)v.x;
  xh[i + 1] = (_Float16)v.y;
  xh[i + 2] = (_Float16)v.z;
  xh[i + 3] = (_Float16)v.w;
}

// ---- 2) weight transpose+convert: wt[n][k] = (f16)w[k][n], 1024x1024 ----
__global__ void k_wt(const float* __restrict__ w, _Float16* __restrict__ wt) {
  __shared__ float t[32][33];
  const int tx = threadIdx.x, ty = threadIdx.y;
  const int x0 = blockIdx.x * 32 + tx;
  const int y0 = blockIdx.y * 32 + ty;
#pragma unroll
  for (int i = 0; i < 4; ++i) t[ty + i * 8][tx] = w[(size_t)(y0 + i * 8) * 1024 + x0];
  __syncthreads();
  const int xo = blockIdx.y * 32 + tx;
  const int yo = blockIdx.x * 32 + ty;
#pragma unroll
  for (int i = 0; i < 4; ++i) wt[(size_t)(yo + i * 8) * 1024 + xo] = (_Float16)t[tx][ty + i * 8];
}

// ---- 3) fused QKV projection: one wave = 16x64 output tile of one of {Q,K,V} ----
__global__ void k_qkv(const _Float16* __restrict__ xh,
                      const _Float16* __restrict__ wqt, const _Float16* __restrict__ wkt,
                      const _Float16* __restrict__ wvt,
                      const float* __restrict__ bq, const float* __restrict__ bk,
                      const float* __restrict__ bv,
                      _Float16* __restrict__ Q, _Float16* __restrict__ K,
                      _Float16* __restrict__ Vt) {
  const int lane = threadIdx.x & 31;
  const int wave = blockIdx.x * 4 + (threadIdx.x >> 5);
  const int w    = wave / (256 * 16);       // 0=Q 1=K 2=V (wave-uniform)
  const int rem  = wave % (256 * 16);
  const int m0   = (rem >> 4) << 4;
  const int n0   = (rem & 15) << 6;
  const _Float16* wt   = (w == 0) ? wqt : (w == 1) ? wkt : wvt;
  const float*    bias = (w == 0) ? bq  : (w == 1) ? bk  : bv;

  v8f acc0 = {}, acc1 = {}, acc2 = {}, acc3 = {};
  const _Float16* arow = xh + (size_t)m0 * D_;
#pragma unroll 4
  for (int k0 = 0; k0 < D_; k0 += 32) {
    __builtin_prefetch(arow + k0 + 256, 0, 0);
    v16h a  = frag_a(arow + k0, D_, lane);
    v16h b0 = frag_b(wt + (size_t)(n0 +  0) * D_ + k0, D_, lane);
    v16h b1 = frag_b(wt + (size_t)(n0 + 16) * D_ + k0, D_, lane);
    v16h b2 = frag_b(wt + (size_t)(n0 + 32) * D_ + k0, D_, lane);
    v16h b3 = frag_b(wt + (size_t)(n0 + 48) * D_ + k0, D_, lane);
    acc0 = wmma_f16(a, b0, acc0);
    acc1 = wmma_f16(a, b1, acc1);
    acc2 = wmma_f16(a, b2, acc2);
    acc3 = wmma_f16(a, b3, acc3);
  }
  const int hf = lane >> 4, n = lane & 15;
  v8f accs[4] = {acc0, acc1, acc2, acc3};
#pragma unroll
  for (int j = 0; j < 4; ++j) {
    const int c  = n0 + j * 16 + n;
    const float bb = bias[c];
    const int hh = c >> 6, dh = c & 63;
#pragma unroll
    for (int r = 0; r < 8; ++r) {
      const int rg = m0 + r + 8 * hf;
      const int bi = rg >> 11, s = rg & 2047;
      const _Float16 val = (_Float16)(accs[j][r] + bb);
      if (w == 2)
        Vt[(((size_t)bi * H_ + hh) * DH_ + dh) * S_ + s] = val;           // V transposed [b,h,dh,s]
      else {
        _Float16* dst = (w == 0) ? Q : K;
        dst[(((size_t)bi * H_ + hh) * S_ + s) * DH_ + dh] = val;          // [b,h,s,dh]
      }
    }
  }
}

// ---- 4) flash attention: one wave = 16 queries of one (b,h); mask: allow j > i ----
__global__ void k_attn(const _Float16* __restrict__ Q, const _Float16* __restrict__ K,
                       const _Float16* __restrict__ Vt, _Float16* __restrict__ ctx) {
  __shared__ __align__(32) _Float16 P[4][16][32];
  const int lane = threadIdx.x & 31;
  const int wv   = threadIdx.x >> 5;
  const int wave = blockIdx.x * 4 + wv;
  const int qt = wave & 127;
  const int h  = (wave >> 7) & 15;
  const int b  = wave >> 11;
  const int q0 = qt * 16;

  const _Float16* Qp = Q  + (((size_t)b * H_ + h) * S_ + q0) * DH_;
  const _Float16* Kp = K  + (((size_t)b * H_ + h) * S_) * DH_;
  const _Float16* Vp = Vt + (((size_t)b * H_ + h) * DH_) * S_;

  const v16h aq0 = frag_a(Qp +  0, DH_, lane);
  const v16h aq1 = frag_a(Qp + 32, DH_, lane);

  const int hf = lane >> 4, n = lane & 15;
  float rm[8], rl[8];
#pragma unroll
  for (int r = 0; r < 8; ++r) { rm[r] = -3.0e38f; rl[r] = 0.0f; }
  v8f acc0 = {}, acc1 = {}, acc2 = {}, acc3 = {};
  const float sc = 0.125f;  // 1/sqrt(64)

#pragma unroll 1
  for (int kk = 0; kk < S_; kk += 32) {
    v8f s0 = {}, s1 = {};
    s0 = wmma_f16(aq0, frag_b(Kp + (size_t)kk * DH_ +  0, DH_, lane), s0);
    s0 = wmma_f16(aq1, frag_b(Kp + (size_t)kk * DH_ + 32, DH_, lane), s0);
    s1 = wmma_f16(aq0, frag_b(Kp + (size_t)(kk + 16) * DH_ +  0, DH_, lane), s1);
    s1 = wmma_f16(aq1, frag_b(Kp + (size_t)(kk + 16) * DH_ + 32, DH_, lane), s1);

    const int key0 = kk + n, key1 = kk + 16 + n;
    float p0[8], p1[8], al[8];
#pragma unroll
    for (int r = 0; r < 8; ++r) {
      const int qi = q0 + r + 8 * hf;
      float a0 = s0[r] * sc + ((key0 > qi) ? 0.0f : -1.0e9f);
      float a1 = s1[r] * sc + ((key1 > qi) ? 0.0f : -1.0e9f);
      float tm = fmaxf(a0, a1);
      tm = fmaxf(tm, __shfl_xor(tm, 1, 32));
      tm = fmaxf(tm, __shfl_xor(tm, 2, 32));
      tm = fmaxf(tm, __shfl_xor(tm, 4, 32));
      tm = fmaxf(tm, __shfl_xor(tm, 8, 32));
      const float mn = fmaxf(rm[r], tm);
      const float alpha = __expf(rm[r] - mn);
      rm[r] = mn;
      const float e0 = __expf(a0 - mn);
      const float e1 = __expf(a1 - mn);
      float rs = e0 + e1;
      rs += __shfl_xor(rs, 1, 32);
      rs += __shfl_xor(rs, 2, 32);
      rs += __shfl_xor(rs, 4, 32);
      rs += __shfl_xor(rs, 8, 32);
      rl[r] = rl[r] * alpha + rs;
      al[r] = alpha; p0[r] = e0; p1[r] = e1;
    }
#pragma unroll
    for (int r = 0; r < 8; ++r) {
      acc0[r] *= al[r]; acc1[r] *= al[r]; acc2[r] *= al[r]; acc3[r] *= al[r];
      P[wv][r + 8 * hf][n]      = (_Float16)p0[r];
      P[wv][r + 8 * hf][16 + n] = (_Float16)p1[r];
    }
    __syncthreads();  // uniform trip count across all 4 waves
    const v16h pa = frag_a(&P[wv][0][0], 32, lane);
    acc0 = wmma_f16(pa, frag_b(Vp + (size_t) 0 * S_ + kk, S_, lane), acc0);
    acc1 = wmma_f16(pa, frag_b(Vp + (size_t)16 * S_ + kk, S_, lane), acc1);
    acc2 = wmma_f16(pa, frag_b(Vp + (size_t)32 * S_ + kk, S_, lane), acc2);
    acc3 = wmma_f16(pa, frag_b(Vp + (size_t)48 * S_ + kk, S_, lane), acc3);
    __syncthreads();
  }
#pragma unroll
  for (int r = 0; r < 8; ++r) {
    const float inv = 1.0f / rl[r];
    const int rg = q0 + r + 8 * hf;
    const size_t rowoff = ((size_t)b * S_ + rg) * D_ + (size_t)h * DH_;
    ctx[rowoff +  0 + n] = (_Float16)(acc0[r] * inv);
    ctx[rowoff + 16 + n] = (_Float16)(acc1[r] * inv);
    ctx[rowoff + 32 + n] = (_Float16)(acc2[r] * inv);
    ctx[rowoff + 48 + n] = (_Float16)(acc3[r] * inv);
  }
}

// ---- 5) output projection + bias + residual -> y (f32) ----
__global__ void k_out(const _Float16* __restrict__ ctx, const _Float16* __restrict__ wot,
                      const float* __restrict__ bo, const float* __restrict__ x,
                      float* __restrict__ y) {
  const int lane = threadIdx.x & 31;
  const int wave = blockIdx.x * 4 + (threadIdx.x >> 5);
  const int m0 = (wave >> 4) << 4;
  const int n0 = (wave & 15) << 6;
  v8f acc0 = {}, acc1 = {}, acc2 = {}, acc3 = {};
  const _Float16* arow = ctx + (size_t)m0 * D_;
#pragma unroll 4
  for (int k0 = 0; k0 < D_; k0 += 32) {
    __builtin_prefetch(arow + k0 + 256, 0, 0);
    v16h a  = frag_a(arow + k0, D_, lane);
    v16h b0 = frag_b(wot + (size_t)(n0 +  0) * D_ + k0, D_, lane);
    v16h b1 = frag_b(wot + (size_t)(n0 + 16) * D_ + k0, D_, lane);
    v16h b2 = frag_b(wot + (size_t)(n0 + 32) * D_ + k0, D_, lane);
    v16h b3 = frag_b(wot + (size_t)(n0 + 48) * D_ + k0, D_, lane);
    acc0 = wmma_f16(a, b0, acc0);
    acc1 = wmma_f16(a, b1, acc1);
    acc2 = wmma_f16(a, b2, acc2);
    acc3 = wmma_f16(a, b3, acc3);
  }
  const int hf = lane >> 4, n = lane & 15;
  v8f accs[4] = {acc0, acc1, acc2, acc3};
#pragma unroll
  for (int j = 0; j < 4; ++j) {
    const int c = n0 + j * 16 + n;
    const float bb = bo[c];
#pragma unroll
    for (int r = 0; r < 8; ++r) {
      const int rg = m0 + r + 8 * hf;
      y[(size_t)rg * D_ + c] = accs[j][r] + bb + x[(size_t)rg * D_ + c];
    }
  }
}

// ---- 6) LayerNorm per row (two-pass like the reference) ----
__global__ void k_ln(const float* __restrict__ y, const float* __restrict__ gamma,
                     const float* __restrict__ beta, float* __restrict__ out) {
  __shared__ float red[8];
  const int row = blockIdx.x;
  const int t = threadIdx.x;
  const float* yr = y + (size_t)row * D_;
  float v[4];
#pragma unroll
  for (int i = 0; i < 4; ++i) v[i] = yr[t + i * 256];
  float s = v[0] + v[1] + v[2] + v[3];
#pragma unroll
  for (int m = 16; m >= 1; m >>= 1) s += __shfl_xor(s, m, 32);
  if ((t & 31) == 0) red[t >> 5] = s;
  __syncthreads();
  float mu = 0.0f;
#pragma unroll
  for (int i = 0; i < 8; ++i) mu += red[i];
  mu *= (1.0f / 1024.0f);
  __syncthreads();
  float s2 = 0.0f;
#pragma unroll
  for (int i = 0; i < 4; ++i) { const float d = v[i] - mu; s2 += d * d; }
#pragma unroll
  for (int m = 16; m >= 1; m >>= 1) s2 += __shfl_xor(s2, m, 32);
  if ((t & 31) == 0) red[t >> 5] = s2;
  __syncthreads();
  float var = 0.0f;
#pragma unroll
  for (int i = 0; i < 8; ++i) var += red[i];
  var *= (1.0f / 1024.0f);
  const float rs = rsqrtf(var + 1e-6f);
#pragma unroll
  for (int i = 0; i < 4; ++i) {
    const int c = t + i * 256;
    out[(size_t)row * D_ + c] = (v[i] - mu) * rs * gamma[c] + beta[c];
  }
}

extern "C" void kernel_launch(void* const* d_in, const int* in_sizes, int n_in,
                              void* d_out, int out_size, void* d_ws, size_t ws_size,
                              hipStream_t stream) {
  const float* x     = (const float*)d_in[0];
  const float* Wq    = (const float*)d_in[1];
  const float* bq    = (const float*)d_in[2];
  const float* Wk    = (const float*)d_in[3];
  const float* bk    = (const float*)d_in[4];
  const float* Wv    = (const float*)d_in[5];
  const float* bv    = (const float*)d_in[6];
  const float* Wo    = (const float*)d_in[7];
  const float* bo    = (const float*)d_in[8];
  const float* gamma = (const float*)d_in[9];
  const float* beta  = (const float*)d_in[10];
  float* out = (float*)d_out;

  char* ws = (char*)d_ws;
  // offsets (bytes); y reuses the xh region (xh dead after k_qkv)
  _Float16* xh  = (_Float16*)(ws + 0);           // 16,777,216 B (also 'y' later)
  float*    y   = (float*)   (ws + 0);
  _Float16* wqt = (_Float16*)(ws + 16777216);    // 2 MB each
  _Float16* wkt = (_Float16*)(ws + 18874368);
  _Float16* wvt = (_Float16*)(ws + 20971520);
  _Float16* wot = (_Float16*)(ws + 23068672);
  _Float16* Qb  = (_Float16*)(ws + 25165824);    // 8,388,608 B each
  _Float16* Kb  = (_Float16*)(ws + 33554432);
  _Float16* Vtb = (_Float16*)(ws + 41943040);
  _Float16* ctx = (_Float16*)(ws + 50331648);    // end = 58,720,256 B

  k_cvt_x<<<8192, 256, 0, stream>>>(x, xh);
  dim3 tb(32, 8), tg(32, 32);
  k_wt<<<tg, tb, 0, stream>>>(Wq, wqt);
  k_wt<<<tg, tb, 0, stream>>>(Wk, wkt);
  k_wt<<<tg, tb, 0, stream>>>(Wv, wvt);
  k_wt<<<tg, tb, 0, stream>>>(Wo, wot);
  k_qkv<<<3072, 128, 0, stream>>>(xh, wqt, wkt, wvt, bq, bk, bv, Qb, Kb, Vtb);
  k_attn<<<1024, 128, 0, stream>>>(Qb, Kb, Vtb, ctx);
  k_out<<<1024, 128, 0, stream>>>(ctx, wot, bo, x, y);
  k_ln<<<4096, 256, 0, stream>>>(y, gamma, beta, out);
}
